// JiTAttention_90769838833766
// MI455X (gfx1250) — matrix operand order repaired and checked by reading, not compiled, not run on hardware
//
#include <hip/hip_runtime.h>

#define DIM    1024
#define NHEADS 16
#define HDIM   64
#define BATCH  4
#define SEQ    2048
#define NTOK   (BATCH * SEQ)     /* 8192 */
#define QKV_N  (3 * DIM)         /* 3072 */

typedef __attribute__((ext_vector_type(16))) __bf16 v16bf;
typedef __attribute__((ext_vector_type(8)))  float  v8f;
typedef unsigned short ushort_t;
typedef unsigned int   uint_t;

// ---------- bf16 helpers (RNE) ----------
__device__ __forceinline__ ushort_t f2bf(float f) {
  uint_t u = __float_as_uint(f);
  u += 0x7FFFu + ((u >> 16) & 1u);
  return (ushort_t)(u >> 16);
}
__device__ __forceinline__ __bf16 bits2bf(ushort_t s) {
  union { ushort_t u; __bf16 b; } c; c.u = s; return c.b;
}
__device__ __forceinline__ void put2(v16bf& v, int j, uint_t dw) {
  union { uint_t u; ushort_t s[2]; } c; c.u = dw;
  v[2 * j]     = bits2bf(c.s[0]);
  v[2 * j + 1] = bits2bf(c.s[1]);
}

// ---------- WMMA wrapper ----------
__device__ __forceinline__ v8f wmma_bf16(v16bf a, v16bf b, v8f c) {
  return __builtin_amdgcn_wmma_f32_16x16x32_bf16(false, a, false, b, (short)0, c,
                                                 false, false);
}

// ---------- fragment loaders (layouts per ISA 7.12.2) ----------
// A 16x32 bf16: lane row m = lane%16; VGPR j holds K = (j<4?0:16) + (lane/16)*8 + (j&3)*2
__device__ __forceinline__ v16bf load_a16x32(const ushort_t* base, int ld) {
  int lane = threadIdx.x & 31, m = lane & 15, hh = lane >> 4;
  v16bf a;
#pragma unroll
  for (int j = 0; j < 8; ++j) {
    int k = ((j & 4) ? 16 : 0) + hh * 8 + (j & 3) * 2;
    put2(a, j, *(const uint_t*)(base + (size_t)m * ld + k));
  }
  return a;
}
// B 32x16 bf16, memory holds element(K,N) at base[N*ld + K] (K contiguous pairs)
__device__ __forceinline__ v16bf load_b16(const ushort_t* base, int ld) {
  int lane = threadIdx.x & 31, n = lane & 15, hh = lane >> 4;
  v16bf b;
#pragma unroll
  for (int j = 0; j < 8; ++j) {
    int k = ((j & 4) ? 16 : 0) + hh * 8 + (j & 3) * 2;
    put2(b, j, *(const uint_t*)(base + (size_t)n * ld + k));
  }
  return b;
}

// ---------- kernel 0: fp32 -> bf16 ----------
__global__ __launch_bounds__(256) void cvt_bf16(const float* __restrict__ in,
                                                ushort_t* __restrict__ out, int n) {
  int i = blockIdx.x * 256 + threadIdx.x;
  if (i < n) out[i] = f2bf(in[i]);
}

// ---------- kernel 1: QKV GEMM, 32x64 tile / wave (2x4 WMMA blocking) ----------
// out[m,n] = sum_k xb[m,k] * wb[n,k] + bias[n]
// n <  2048 : q/k  -> fp32 qk_out[m, n]
// n >= 2048 : v    -> bf16 TRANSPOSED vt[((b*H + h)*64 + d) * SEQ + s]
__global__ __launch_bounds__(128) void gemm_qkv(const ushort_t* __restrict__ A,
                                                const ushort_t* __restrict__ W,
                                                const float* __restrict__ bias,
                                                float* __restrict__ qk_out,
                                                ushort_t* __restrict__ vt_out) {
  const int K = DIM, N = QKV_N;
  int wid = blockIdx.x * 4 + (threadIdx.x >> 5);
  int nsup = N >> 6;                         // 48 super-tiles of 64 cols
  int m0 = (wid / nsup) << 5, n0 = (wid % nsup) << 6;
  v8f acc[2][4];
#pragma unroll
  for (int i = 0; i < 2; ++i)
#pragma unroll
    for (int j = 0; j < 4; ++j) acc[i][j] = (v8f){};

  for (int k0 = 0; k0 < K; k0 += 32) {
    v16bf a[2], b[4];
#pragma unroll
    for (int i = 0; i < 2; ++i)
      a[i] = load_a16x32(A + (size_t)(m0 + 16 * i) * K + k0, K);
#pragma unroll
    for (int j = 0; j < 4; ++j)
      b[j] = load_b16(W + (size_t)(n0 + 16 * j) * K + k0, K);
#pragma unroll
    for (int i = 0; i < 2; ++i)
#pragma unroll
      for (int j = 0; j < 4; ++j) acc[i][j] = wmma_bf16(a[i], b[j], acc[i][j]);
  }

  int lane = threadIdx.x & 31, hh = lane >> 4, nn = lane & 15;
#pragma unroll
  for (int j = 0; j < 4; ++j) {
    int n = n0 + 16 * j + nn;
    float bn = bias[n];
#pragma unroll
    for (int i = 0; i < 2; ++i) {
#pragma unroll
      for (int r = 0; r < 8; ++r) {
        int m = m0 + 16 * i + hh * 8 + r;
        float v = acc[i][j][r] + bn;
        if (n < 2 * DIM) {
          qk_out[(size_t)m * (2 * DIM) + n] = v;
        } else {
          int dc = n - 2 * DIM;                        // h*64 + d
          int b_ = m >> 11, s_ = m & (SEQ - 1);        // token -> (b, s)
          vt_out[((size_t)b_ * NHEADS * HDIM + dc) * SEQ + s_] = f2bf(v);
        }
      }
    }
  }
}

// ---------- kernel 2: RMSNorm + RoPE on q,k (one wave per token-head) ----------
__global__ __launch_bounds__(256) void normrope(const float* __restrict__ qk,
                                                const float* __restrict__ cosb,
                                                const float* __restrict__ sinb,
                                                const float* __restrict__ qw,
                                                const float* __restrict__ kw,
                                                ushort_t* __restrict__ qo,
                                                ushort_t* __restrict__ ko) {
  int job = blockIdx.x * 8 + (threadIdx.x >> 5);  // 2 * NTOK * NHEADS jobs
  int lane = threadIdx.x & 31;
  int which = (job >= NTOK * NHEADS) ? 1 : 0;
  int rem = job - which * NTOK * NHEADS;
  int t = rem >> 4, h = rem & 15;
  int s = t & (SEQ - 1);
  const float* src = qk + (size_t)t * (2 * DIM) + which * DIM + h * HDIM;
  const float* w   = which ? kw : qw;
  ushort_t*    dst = (which ? ko : qo) + (size_t)t * DIM + h * HDIM;
  int d0 = 2 * lane;
  float a = src[d0], b = src[d0 + 1];
  float ss = a * a + b * b;
#pragma unroll
  for (int mask = 1; mask < 32; mask <<= 1) ss += __shfl_xor(ss, mask, 32);
  float inv = rsqrtf(ss * (1.0f / HDIM) + 1e-6f);
  a = a * inv * w[d0];
  b = b * inv * w[d0 + 1];
  float c0 = cosb[(size_t)s * HDIM + d0],     s0 = sinb[(size_t)s * HDIM + d0];
  float c1 = cosb[(size_t)s * HDIM + d0 + 1], s1 = sinb[(size_t)s * HDIM + d0 + 1];
  dst[d0]     = f2bf(a * c0 - b * s0);
  dst[d0 + 1] = f2bf(b * c1 + a * s1);
}

// ---------- kernel 3: flash attention, 32 queries / wave ----------
// Q, K: bf16 [B*S, H*64] row-major.  Vt: bf16 [B, H, 64, SEQ].  O: bf16 [B*S, H*64].
__global__ __launch_bounds__(128) void attn_fa(const ushort_t* __restrict__ Q,
                                               const ushort_t* __restrict__ Kt,
                                               const ushort_t* __restrict__ Vt,
                                               ushort_t* __restrict__ O) {
  __shared__ __align__(16) ushort_t Plds[4][2][16 * 32];
  int wv = threadIdx.x >> 5;
  int job = blockIdx.x * 4 + wv;              // B*H*(S/32) = 4096 jobs
  int qt = job & (SEQ / 32 - 1);              // 64 query super-tiles
  int bh = job >> 6, h = bh & 15, b = bh >> 4;
  const size_t tokbase = (size_t)b * SEQ;
  int lane = threadIdx.x & 31, nn = lane & 15, hh = lane >> 4;
  const int qrow = qt * 32;

  v16bf qa[2][2];
#pragma unroll
  for (int q = 0; q < 2; ++q) {
    const ushort_t* qp = Q + (tokbase + qrow + q * 16) * DIM + h * HDIM;
    qa[q][0] = load_a16x32(qp, DIM);
    qa[q][1] = load_a16x32(qp + 32, DIM);
  }

  float rm[2][8], rl[2][8];
  v8f o[2][4];
#pragma unroll
  for (int q = 0; q < 2; ++q) {
#pragma unroll
    for (int r = 0; r < 8; ++r) { rm[q][r] = -1e30f; rl[q][r] = 0.0f; }
#pragma unroll
    for (int f = 0; f < 4; ++f) o[q][f] = (v8f){};
  }

  const ushort_t* vbase = Vt + ((size_t)bh * HDIM) * SEQ;
  const float sc = 0.125f;  // 1/sqrt(64)

  for (int kb = 0; kb < SEQ; kb += 32) {
    const ushort_t* kp = Kt + (tokbase + kb) * DIM + h * HDIM;
    v16bf kf[2][2];                           // [key col group][d half]
    kf[0][0] = load_b16(kp, DIM);
    kf[0][1] = load_b16(kp + 32, DIM);
    kf[1][0] = load_b16(kp + 16 * DIM, DIM);
    kf[1][1] = load_b16(kp + 16 * DIM + 32, DIM);

    __asm volatile("s_wait_dscnt 0" ::: "memory");  // WAR vs prior P reads
#pragma unroll
    for (int q = 0; q < 2; ++q) {
      v8f s0 = {}, s1 = {};
      s0 = wmma_bf16(qa[q][0], kf[0][0], s0);
      s0 = wmma_bf16(qa[q][1], kf[0][1], s0);
      s1 = wmma_bf16(qa[q][0], kf[1][0], s1);
      s1 = wmma_bf16(qa[q][1], kf[1][1], s1);

      ushort_t* pl = &Plds[wv][q][0];
      float fac[8];
#pragma unroll
      for (int r = 0; r < 8; ++r) {
        float lm = fmaxf(s0[r], s1[r]) * sc;
#pragma unroll
        for (int mask = 1; mask < 16; mask <<= 1)
          lm = fmaxf(lm, __shfl_xor(lm, mask, 32));
        float mn = fmaxf(rm[q][r], lm);
        fac[r] = __expf(rm[q][r] - mn);
        float e0 = __expf(s0[r] * sc - mn);
        float e1 = __expf(s1[r] * sc - mn);
        float ls = e0 + e1;
#pragma unroll
        for (int mask = 1; mask < 16; mask <<= 1) ls += __shfl_xor(ls, mask, 32);
        rl[q][r] = rl[q][r] * fac[r] + ls;
        rm[q][r] = mn;
        int row = hh * 8 + r;
        pl[row * 32 + nn]      = f2bf(e0);
        pl[row * 32 + 16 + nn] = f2bf(e1);
      }
#pragma unroll
      for (int f = 0; f < 4; ++f)
#pragma unroll
        for (int r = 0; r < 8; ++r) o[q][f][r] *= fac[r];
    }

    __asm volatile("s_wait_dscnt 0" ::: "memory");  // transpose visible wave-wide
    v16bf pa0 = load_a16x32(&Plds[wv][0][0], 32);
    v16bf pa1 = load_a16x32(&Plds[wv][1][0], 32);
#pragma unroll
    for (int f = 0; f < 4; ++f) {
      v16bf vf = load_b16(vbase + (size_t)(f * 16) * SEQ + kb, SEQ);
      o[0][f] = wmma_bf16(pa0, vf, o[0][f]);
      o[1][f] = wmma_bf16(pa1, vf, o[1][f]);
    }
  }

#pragma unroll
  for (int q = 0; q < 2; ++q) {
    ushort_t* op = O + (tokbase + qrow + q * 16) * DIM + h * HDIM;
#pragma unroll
    for (int r = 0; r < 8; ++r) {
      float inv = 1.0f / rl[q][r];
      size_t m = (size_t)(hh * 8 + r) * DIM;
#pragma unroll
      for (int f = 0; f < 4; ++f)
        op[m + f * 16 + nn] = f2bf(o[q][f][r] * inv);
    }
  }
}

// ---------- kernel 4: output projection GEMM, 32x64 tile / wave ----------
__global__ __launch_bounds__(128) void gemm_proj(const ushort_t* __restrict__ A,
                                                 const ushort_t* __restrict__ W,
                                                 const float* __restrict__ bias,
                                                 float* __restrict__ out) {
  const int K = DIM, N = DIM;
  int wid = blockIdx.x * 4 + (threadIdx.x >> 5);
  int nsup = N >> 6;                          // 16
  int m0 = (wid / nsup) << 5, n0 = (wid % nsup) << 6;
  v8f acc[2][4];
#pragma unroll
  for (int i = 0; i < 2; ++i)
#pragma unroll
    for (int j = 0; j < 4; ++j) acc[i][j] = (v8f){};

  for (int k0 = 0; k0 < K; k0 += 32) {
    v16bf a[2], b[4];
#pragma unroll
    for (int i = 0; i < 2; ++i)
      a[i] = load_a16x32(A + (size_t)(m0 + 16 * i) * K + k0, K);
#pragma unroll
    for (int j = 0; j < 4; ++j)
      b[j] = load_b16(W + (size_t)(n0 + 16 * j) * K + k0, K);
#pragma unroll
    for (int i = 0; i < 2; ++i)
#pragma unroll
      for (int j = 0; j < 4; ++j) acc[i][j] = wmma_bf16(a[i], b[j], acc[i][j]);
  }

  int lane = threadIdx.x & 31, hh = lane >> 4, nn = lane & 15;
#pragma unroll
  for (int j = 0; j < 4; ++j) {
    int n = n0 + 16 * j + nn;
    float bn = bias[n];
#pragma unroll
    for (int i = 0; i < 2; ++i)
#pragma unroll
      for (int r = 0; r < 8; ++r) {
        int m = m0 + 16 * i + hh * 8 + r;
        out[(size_t)m * N + n] = acc[i][j][r] + bn;
      }
  }
}

extern "C" void kernel_launch(void* const* d_in, const int* in_sizes, int n_in,
                              void* d_out, int out_size, void* d_ws, size_t ws_size,
                              hipStream_t stream) {
  const float* x       = (const float*)d_in[0];
  const float* ropec   = (const float*)d_in[1];
  const float* ropes   = (const float*)d_in[2];
  const float* qkv_w   = (const float*)d_in[3];
  const float* qkv_b   = (const float*)d_in[4];
  const float* proj_w  = (const float*)d_in[5];
  const float* proj_b  = (const float*)d_in[6];
  const float* q_nw    = (const float*)d_in[7];
  const float* k_nw    = (const float*)d_in[8];
  float* out = (float*)d_out;

  // workspace layout (bytes)
  char* ws = (char*)d_ws;
  ushort_t* xb     = (ushort_t*)(ws);                 // 16 MiB
  ushort_t* wqkvb  = (ushort_t*)(ws + 16777216);      //  6 MiB
  ushort_t* wprojb = (ushort_t*)(ws + 23068672);      //  2 MiB
  float*    qkf32  = (float*)   (ws + 25165824);      // 64 MiB
  ushort_t* qb     = (ushort_t*)(ws + 92274688);      // 16 MiB
  ushort_t* kb     = (ushort_t*)(ws + 109051904);     // 16 MiB
  ushort_t* vt     = (ushort_t*)(ws + 125829120);     // 16 MiB (transposed V)
  ushort_t* ab     = (ushort_t*)(ws + 142606336);     // 16 MiB

  // 0) fp32 -> bf16 conversions
  {
    int n = NTOK * DIM;
    cvt_bf16<<<(n + 255) / 256, 256, 0, stream>>>(x, xb, n);
    n = QKV_N * DIM;
    cvt_bf16<<<(n + 255) / 256, 256, 0, stream>>>(qkv_w, wqkvb, n);
    n = DIM * DIM;
    cvt_bf16<<<(n + 255) / 256, 256, 0, stream>>>(proj_w, wprojb, n);
  }
  // 1) QKV GEMM: (256 m-supertiles) x (48 n-supertiles) = 12288 waves / 4 per block
  gemm_qkv<<<3072, 128, 0, stream>>>(xb, wqkvb, qkv_b, qkf32, vt);
  // 2) RMSNorm + RoPE: 262144 wave-jobs / 8 per block
  normrope<<<32768, 256, 0, stream>>>(qkf32, ropec, ropes, q_nw, k_nw, qb, kb);
  // 3) flash attention: 4*16*64 = 4096 wave-jobs / 4 per block
  attn_fa<<<1024, 128, 0, stream>>>(qb, kb, vt, ab);
  // 4) output projection: 256 x 16 = 4096 waves / 4 per block
  gemm_proj<<<1024, 128, 0, stream>>>(ab, wprojb, proj_b, out);
}
